// ViT_63101659512861
// MI455X (gfx1250) — matrix operand order repaired and checked by reading, not compiled, not run on hardware
//
#include <hip/hip_runtime.h>

// ---------------- types ----------------
typedef _Float16 f16_t;
typedef __attribute__((ext_vector_type(16))) _Float16 v16h;
typedef __attribute__((ext_vector_type(8)))  _Float16 v8h;
typedef __attribute__((ext_vector_type(8)))  float    v8f;

#define BB 32
#define DD 768
#define LL 12
#define TT_ 197
#define TPAD 224
#define NCC 1000
#define PSQ 256           // 16*16 patch dim
#define NPATCH 196
#define BT (BB*TT_)       // 6304
#define M196 (BB*NPATCH)  // 6272

// GEMM epilogue flags
#define F_BIAS 1
#define F_GELU 2
#define F_F32  4
#define F_F16  8

// ---------------- f32 -> f16 elementwise ----------------
__global__ void k_cvt16(const float* __restrict__ in, f16_t* __restrict__ out, int n) {
    int i = blockIdx.x * blockDim.x + threadIdx.x;
    if (i < n) out[i] = (f16_t)in[i];
}

// ---------------- tiled transpose f32[R,C] -> f16[C, ldo] (zero pad rows R..ldo) ----
__global__ void k_transpose_pad_f16(const float* __restrict__ in, long sIn,
                                    f16_t* __restrict__ out, long sOut,
                                    int R, int C, int ldo) {
    __shared__ float tile[32][33];
    const float* ib = in + (size_t)blockIdx.z * sIn;
    f16_t* ob = out + (size_t)blockIdx.z * sOut;
    int r0 = blockIdx.y * 32, c0 = blockIdx.x * 32;
    for (int i = threadIdx.y; i < 32; i += 8) {
        int r = r0 + i, c = c0 + threadIdx.x;
        tile[i][threadIdx.x] = (r < R && c < C) ? ib[(size_t)r * C + c] : 0.f;
    }
    __syncthreads();
    for (int i = threadIdx.y; i < 32; i += 8) {
        int c = c0 + i, r = r0 + threadIdx.x;
        if (c < C && r < ldo) ob[(size_t)c * ldo + r] = (f16_t)tile[threadIdx.x][i];
    }
}

// ---------------- generic WMMA GEMM, 64x64 tile per wave (4x4 register blocking) ----
// Out = alpha*(A @ Bt^T) [+bias][gelu]
// A: [M,K] row-major, row stride lda. Bt: [N,K] row-major (i.e. B transposed).
// Each 32-thread wave computes a 64x64 output tile: 16 v8f accumulators,
// 4 A + 4 B fragments per K-step -> 16 v_wmma_f32_16x16x32_f16 per 16 b128 loads
// (1 WMMA per load; 4x operand reuse out of registers).
__global__ void __launch_bounds__(32)
k_wmma_gemm(const f16_t* __restrict__ A, long sA, int lda,
            const f16_t* __restrict__ Bt, long sB,
            const float* __restrict__ bias,
            float* __restrict__ Out, long sO, int ldo,
            f16_t* __restrict__ Out16, long sO16, int ldo16,
            int M, int N, int K, float alpha, int flags) {
    const int lane = threadIdx.x;
    const int hi = lane >> 4, ln = lane & 15;
    const int bz = blockIdx.z;
    const f16_t* Ab = A + (size_t)bz * sA;
    const f16_t* Bb = Bt + (size_t)bz * sB;

    const f16_t* ap[4];
    const f16_t* bp[4];
#pragma unroll
    for (int i = 0; i < 4; ++i) {
        int m = blockIdx.x * 64 + i * 16 + ln;      // clamp edge reads; stores predicated
        if (m > M - 1) m = M - 1;
        ap[i] = Ab + (size_t)m * lda + hi * 8;
        int n = blockIdx.y * 64 + i * 16 + ln;
        if (n > N - 1) n = N - 1;
        bp[i] = Bb + (size_t)n * K + hi * 8;
    }

    v8f acc[4][4];
#pragma unroll
    for (int i = 0; i < 4; ++i)
#pragma unroll
        for (int j = 0; j < 4; ++j)
            acc[i][j] = (v8f){0.f,0.f,0.f,0.f,0.f,0.f,0.f,0.f};

    union frag { v16h v; v8h h[2]; };
    for (int k0 = 0; k0 < K; k0 += 32) {
        // speculative prefetch 4 K-steps (128 elements) ahead (OOB prefetches silently dropped)
#pragma unroll
        for (int i = 0; i < 4; ++i) {
            __builtin_prefetch(ap[i] + k0 + 128, 0, 3);
            __builtin_prefetch(bp[i] + k0 + 128, 0, 3);
        }
        frag a[4], b[4];
#pragma unroll
        for (int i = 0; i < 4; ++i) {
            // A fragment: lane holds row m; halves at K = k0+hi*8 and k0+16+hi*8
            a[i].h[0] = *(const v8h*)(ap[i] + k0);
            a[i].h[1] = *(const v8h*)(ap[i] + k0 + 16);
            // B fragment: lane holds col n; halves at K = k0+hi*8 and k0+16+hi*8
            b[i].h[0] = *(const v8h*)(bp[i] + k0);
            b[i].h[1] = *(const v8h*)(bp[i] + k0 + 16);
        }
#pragma unroll
        for (int i = 0; i < 4; ++i)
#pragma unroll
            for (int j = 0; j < 4; ++j)
                acc[i][j] = __builtin_amdgcn_wmma_f32_16x16x32_f16(
                    false, a[i].v, false, b[j].v, (short)0, acc[i][j], false, false);
    }

#pragma unroll
    for (int j = 0; j < 4; ++j) {
        int nc = blockIdx.y * 64 + j * 16 + ln;
        if (nc >= N) continue;
        float bvj = (flags & F_BIAS) ? bias[nc] : 0.f;
#pragma unroll
        for (int i = 0; i < 4; ++i) {
#pragma unroll
            for (int r = 0; r < 8; ++r) {
                int mc = blockIdx.x * 64 + i * 16 + r + 8 * hi;  // C/D: m = r + 8*(lane/16)
                if (mc < M) {
                    float v = acc[i][j][r] * alpha + bvj;
                    if (flags & F_GELU) v = 0.5f * v * (1.f + erff(v * 0.70710678118654752f));
                    if (flags & F_F32)  Out[(size_t)bz * sO + (size_t)mc * ldo + nc] = v;
                    if (flags & F_F16)  Out16[(size_t)bz * sO16 + (size_t)mc * ldo16 + nc] = (f16_t)v;
                }
            }
        }
    }
}

// ---------------- cls token + positional embedding, produce f32 + f16 -----------
__global__ void k_pos_cls(float* __restrict__ hdn, f16_t* __restrict__ h16,
                          const float* __restrict__ cls, const float* __restrict__ pos) {
    int i = blockIdx.x * blockDim.x + threadIdx.x;
    if (i >= BT * DD) return;
    int d = i % DD;
    int t = (i / DD) % TT_;
    float v = (t == 0) ? (cls[d] + pos[d]) : (hdn[i] + pos[(size_t)t * DD + d]);
    hdn[i] = v;
    h16[i] = (f16_t)v;
}

// ---------------- softmax rows of 197, write f16 padded to 224 ------------------
__global__ void k_softmax(const float* __restrict__ att, f16_t* __restrict__ att16, int rows) {
    int row  = blockIdx.x * (blockDim.x / 32) + (threadIdx.x >> 5);
    int lane = threadIdx.x & 31;
    if (row >= rows) return;
    const float* p = att + (size_t)row * TPAD;
    f16_t* o = att16 + (size_t)row * TPAD;
    float mx = -3.0e38f;
    for (int s = lane; s < TT_; s += 32) mx = fmaxf(mx, p[s]);
    for (int m = 16; m; m >>= 1) mx = fmaxf(mx, __shfl_xor(mx, m, 32));
    float e[7]; int cnt = 0; float sum = 0.f;
    for (int s = lane; s < TT_; s += 32) { float t = expf(p[s] - mx); e[cnt++] = t; sum += t; }
    for (int m = 16; m; m >>= 1) sum += __shfl_xor(sum, m, 32);
    float inv = 1.f / sum;
    cnt = 0;
    for (int s = lane; s < TT_; s += 32) o[s] = (f16_t)(e[cnt++] * inv);
    for (int s = TT_ + lane; s < TPAD; s += 32) o[s] = (f16_t)0.f;
}

// ---------------- LayerNorm(h + delta) -> h (f32) and h16 (f16), wave per row ---
__global__ void k_ln_res(float* __restrict__ h, f16_t* __restrict__ h16,
                         const float* __restrict__ delta,
                         const float* __restrict__ g, const float* __restrict__ b,
                         int rows) {
    int row  = blockIdx.x * (blockDim.x / 32) + (threadIdx.x >> 5);
    int lane = threadIdx.x & 31;
    if (row >= rows) return;
    float* hp = h + (size_t)row * DD;
    const float* dp = delta + (size_t)row * DD;
    float x[24];
    float s = 0.f;
#pragma unroll
    for (int i = 0; i < 24; ++i) { x[i] = hp[lane + 32 * i] + dp[lane + 32 * i]; s += x[i]; }
    for (int m = 16; m; m >>= 1) s += __shfl_xor(s, m, 32);
    float mean = s * (1.f / DD);
    float vs = 0.f;
#pragma unroll
    for (int i = 0; i < 24; ++i) { float d = x[i] - mean; vs += d * d; }
    for (int m = 16; m; m >>= 1) vs += __shfl_xor(vs, m, 32);
    float inv = rsqrtf(vs * (1.f / DD) + 1e-5f);
#pragma unroll
    for (int i = 0; i < 24; ++i) {
        int c = lane + 32 * i;
        float y = (x[i] - mean) * inv * g[c] + b[c];
        hp[c] = y;
        h16[(size_t)row * DD + c] = (f16_t)y;
    }
}

// ---------------- final LN of cls tokens -> f16 [32,768] ------------------------
__global__ void k_cls_ln(const float* __restrict__ h, const float* __restrict__ fg,
                         const float* __restrict__ fb, f16_t* __restrict__ out16) {
    int bq = blockIdx.x;          // batch
    int lane = threadIdx.x;       // 32 lanes
    const float* hp = h + (size_t)bq * TT_ * DD;   // token 0 row
    float x[24];
    float s = 0.f;
#pragma unroll
    for (int i = 0; i < 24; ++i) { x[i] = hp[lane + 32 * i]; s += x[i]; }
    for (int m = 16; m; m >>= 1) s += __shfl_xor(s, m, 32);
    float mean = s * (1.f / DD);
    float vs = 0.f;
#pragma unroll
    for (int i = 0; i < 24; ++i) { float d = x[i] - mean; vs += d * d; }
    for (int m = 16; m; m >>= 1) vs += __shfl_xor(vs, m, 32);
    float inv = rsqrtf(vs * (1.f / DD) + 1e-5f);
#pragma unroll
    for (int i = 0; i < 24; ++i) {
        int c = lane + 32 * i;
        out16[(size_t)bq * DD + c] = (f16_t)((x[i] - mean) * inv * fg[c] + fb[c]);
    }
}

// =================================================================================
extern "C" void kernel_launch(void* const* d_in, const int* in_sizes, int n_in,
                              void* d_out, int out_size, void* d_ws, size_t ws_size,
                              hipStream_t stream) {
    (void)in_sizes; (void)n_in; (void)out_size; (void)ws_size;
    const float* x      = (const float*)d_in[0];
    const float* cls    = (const float*)d_in[1];
    const float* emb_w  = (const float*)d_in[2];
    const float* emb_b  = (const float*)d_in[3];
    const float* pos    = (const float*)d_in[4];
    const float* Wq = (const float*)d_in[5],  *bq = (const float*)d_in[6];
    const float* Wk = (const float*)d_in[7],  *bk = (const float*)d_in[8];
    const float* Wv = (const float*)d_in[9],  *bv = (const float*)d_in[10];
    const float* Wo = (const float*)d_in[11], *bo = (const float*)d_in[12];
    const float* g1 = (const float*)d_in[13], *be1 = (const float*)d_in[14];
    const float* W1 = (const float*)d_in[15], *b1 = (const float*)d_in[16];
    const float* W2 = (const float*)d_in[17], *b2 = (const float*)d_in[18];
    const float* g2 = (const float*)d_in[19], *be2 = (const float*)d_in[20];
    const float* fg = (const float*)d_in[21], *fb = (const float*)d_in[22];
    const float* head_w = (const float*)d_in[23], *head_b = (const float*)d_in[24];
    float* out = (float*)d_out;

    // ---- workspace carve-out (256B aligned slots) ----
    char* base = (char*)d_ws;
    size_t off = 0;
    auto alloc = [&](size_t bytes) -> void* {
        void* p = base + off;
        off = (off + bytes + 255) & ~(size_t)255;
        return p;
    };
    float* hdn  = (float*)alloc((size_t)BT * DD * 4);
    f16_t* h16  = (f16_t*)alloc((size_t)BT * DD * 2);
    float* tmp  = (float*)alloc((size_t)BT * DD * 4);
    f16_t* q16  = (f16_t*)alloc((size_t)BT * DD * 2);
    f16_t* k16  = (f16_t*)alloc((size_t)BT * DD * 2);
    f16_t* g16  = (f16_t*)alloc((size_t)BT * DD * 2);
    f16_t* x16  = (f16_t*)alloc((size_t)M196 * PSQ * 2);
    f16_t* ewt  = (f16_t*)alloc((size_t)DD * PSQ * 2);
    f16_t* wt0  = (f16_t*)alloc((size_t)6 * DD * DD * 2);   // 6 per-layer transposed weights
    f16_t* vt16 = (f16_t*)alloc((size_t)BB * DD * TPAD * 2);
    float* att  = (float*)alloc((size_t)BB * TT_ * TPAD * 4);
    f16_t* att16= (f16_t*)alloc((size_t)BB * TT_ * TPAD * 2);
    f16_t* hwt  = (f16_t*)alloc((size_t)NCC * DD * 2);
    f16_t* c16  = (f16_t*)alloc((size_t)BB * DD * 2);

    const dim3 tb(32, 8, 1);
    const float scale = 0.036084391824351613f;   // 768^-0.5

    // 1) patches f32 -> f16 (x is already [B,196,256] contiguous)
    {
        int n = M196 * PSQ;
        k_cvt16<<<dim3((n + 255) / 256), dim3(256), 0, stream>>>(x, x16, n);
    }
    // 2) emb_w [256,768] -> ewt [768,256]
    k_transpose_pad_f16<<<dim3(24, 8, 1), tb, 0, stream>>>(emb_w, 0, ewt, 0, PSQ, DD, PSQ);
    // 3) patch embed GEMM into hdn rows t>=1 (batched, per-batch row offset of 1 token)
    k_wmma_gemm<<<dim3(4, DD / 64, BB), dim3(32), 0, stream>>>(
        x16, (long)NPATCH * PSQ, PSQ, ewt, 0, emb_b,
        hdn + DD, (long)TT_ * DD, DD, nullptr, 0, 0,
        NPATCH, DD, PSQ, 1.f, F_BIAS | F_F32);
    // 4) cls token + positional embedding; produce h16
    {
        int n = BT * DD;
        k_pos_cls<<<dim3((n + 255) / 256), dim3(256), 0, stream>>>(hdn, h16, cls, pos);
    }

    const int MT = (BT + 63) / 64;   // 99 M-tiles for token-flattened GEMMs

    for (int l = 0; l < LL; ++l) {
        const size_t wOff = (size_t)l * DD * DD;
        const size_t vOff = (size_t)l * DD;
        const float* Wm[6] = { Wq + wOff, Wk + wOff, Wv + wOff, Wo + wOff, W1 + wOff, W2 + wOff };
        // transpose this layer's 6 weight matrices to f16 [N,K]
        for (int j = 0; j < 6; ++j)
            k_transpose_pad_f16<<<dim3(24, 24, 1), tb, 0, stream>>>(
                Wm[j], 0, wt0 + (size_t)j * DD * DD, 0, DD, DD, DD);

        // q, k (f16 out), v (f32 out in tmp)
        k_wmma_gemm<<<dim3(MT, DD / 64, 1), dim3(32), 0, stream>>>(
            h16, 0, DD, wt0 + 0 * (size_t)DD * DD, 0, bq + vOff,
            nullptr, 0, 0, q16, 0, DD, BT, DD, DD, 1.f, F_BIAS | F_F16);
        k_wmma_gemm<<<dim3(MT, DD / 64, 1), dim3(32), 0, stream>>>(
            h16, 0, DD, wt0 + 1 * (size_t)DD * DD, 0, bk + vOff,
            nullptr, 0, 0, k16, 0, DD, BT, DD, DD, 1.f, F_BIAS | F_F16);
        k_wmma_gemm<<<dim3(MT, DD / 64, 1), dim3(32), 0, stream>>>(
            h16, 0, DD, wt0 + 2 * (size_t)DD * DD, 0, bv + vOff,
            tmp, 0, DD, nullptr, 0, 0, BT, DD, DD, 1.f, F_BIAS | F_F32);

        // v [197,768] -> vt16 [768,224] per batch, zero-padded in s
        k_transpose_pad_f16<<<dim3(24, 7, BB), tb, 0, stream>>>(
            tmp, (long)TT_ * DD, vt16, (long)DD * TPAD, TT_, DD, TPAD);

        // scores = q @ k^T * scale (batched)
        k_wmma_gemm<<<dim3(4, 4, BB), dim3(32), 0, stream>>>(
            q16, (long)TT_ * DD, DD, k16, (long)TT_ * DD, nullptr,
            att, (long)TT_ * TPAD, TPAD, nullptr, 0, 0,
            TT_, TT_, DD, scale, F_F32);

        // softmax -> att16 (padded rows of 224)
        k_softmax<<<dim3((BT + 3) / 4), dim3(128), 0, stream>>>(att, att16, BT);

        // att @ v -> g16 (f16)
        k_wmma_gemm<<<dim3(4, DD / 64, BB), dim3(32), 0, stream>>>(
            att16, (long)TT_ * TPAD, TPAD, vt16, (long)DD * TPAD, nullptr,
            nullptr, 0, 0, g16, (long)TT_ * DD, DD,
            TT_, DD, TPAD, 1.f, F_F16);

        // attention output projection -> tmp (f32)
        k_wmma_gemm<<<dim3(MT, DD / 64, 1), dim3(32), 0, stream>>>(
            g16, 0, DD, wt0 + 3 * (size_t)DD * DD, 0, bo + vOff,
            tmp, 0, DD, nullptr, 0, 0, BT, DD, DD, 1.f, F_BIAS | F_F32);

        // h = LN(h + attn_out)
        k_ln_res<<<dim3((BT + 7) / 8), dim3(256), 0, stream>>>(
            hdn, h16, tmp, g1 + vOff, be1 + vOff, BT);

        // MLP: g16 = gelu(h16 @ W1 + b1) ; tmp = g16 @ W2 + b2
        k_wmma_gemm<<<dim3(MT, DD / 64, 1), dim3(32), 0, stream>>>(
            h16, 0, DD, wt0 + 4 * (size_t)DD * DD, 0, b1 + vOff,
            nullptr, 0, 0, g16, 0, DD, BT, DD, DD, 1.f, F_BIAS | F_GELU | F_F16);
        k_wmma_gemm<<<dim3(MT, DD / 64, 1), dim3(32), 0, stream>>>(
            g16, 0, DD, wt0 + 5 * (size_t)DD * DD, 0, b2 + vOff,
            tmp, 0, DD, nullptr, 0, 0, BT, DD, DD, 1.f, F_BIAS | F_F32);

        // h = LN(h + mlp_out)
        k_ln_res<<<dim3((BT + 7) / 8), dim3(256), 0, stream>>>(
            hdn, h16, tmp, g2 + vOff, be2 + vOff, BT);
    }

    // head: transpose head_w [768,1000] -> hwt [1000,768]
    k_transpose_pad_f16<<<dim3(32, 24, 1), tb, 0, stream>>>(head_w, 0, hwt, 0, DD, NCC, DD);
    // final LN of cls tokens -> c16 [32,768]
    k_cls_ln<<<dim3(BB), dim3(32), 0, stream>>>(hdn, fg, fb, c16);
    // out = c16 @ head_w + head_b  -> [32,1000] f32
    k_wmma_gemm<<<dim3(1, (NCC + 63) / 64, 1), dim3(32), 0, stream>>>(
        c16, 0, DD, hwt, 0, head_b,
        out, 0, NCC, nullptr, 0, 0, BB, NCC, DD, 1.f, F_BIAS | F_F32);
}
